// Attention_10857677325079
// MI455X (gfx1250) — matrix operand order repaired and checked by reading, not compiled, not run on hardware
//
#include <hip/hip_runtime.h>

// FlashAttention-2 forward, causal, GQA (rep=4), fp32 I/O, f16 WMMA math.
// S=2048, Hq=32, Hkv=8, D=128.  MI455X / gfx1250, wave32.
//
// Fast path (needs 8MB workspace):
//   1) pre-pass converts K -> f16 [kvh][s][d] and V -> f16 transposed
//      [kvh][d][s] (one-time ~25MB of HBM traffic ~= 1us at 23.3TB/s)
//   2) hot loop stages 64-key K/V tiles into LDS with
//      GLOBAL_LOAD_ASYNC_TO_LDS_B128 (ASYNCcnt, no VGPR round trip, no
//      conversions), then runs 32 v_wmma_f32_16x16x32_f16 per iteration.
// Fallback path (small workspace): in-loop conversion variant.
//
// S^T = K*Q^T and O^T = V^T*P^T keep the softmax q-row lane-private
// (reductions = one ds_swizzle SWAPX16).  Q is pre-scaled by scale*log2e.
// With BR=BC=64 every wave's causal diagonal falls in the last key block.

typedef __attribute__((ext_vector_type(16))) _Float16 v16h;
typedef __attribute__((ext_vector_type(8)))  _Float16 v8h;
typedef __attribute__((ext_vector_type(4)))  _Float16 v4h;
typedef __attribute__((ext_vector_type(8)))  float    v8f;

#define SEQ_LEN   2048
#define NUM_HEADS 32
#define NUM_KV    8
#define HD        128
#define ATT_SCALE 0.08838834764831845f
#define LOG2E     1.4426950408889634f

#define BR 64          // q rows per workgroup (4 waves x 16)
#define BC 64          // keys per inner iteration (4 key tiles of 16)
#define KSTR 136       // LDS row stride (halves) for K tile  [key][d]
#define VSTR 72        // LDS row stride (halves) for V^T tile [d][key]

__device__ __forceinline__ v16h join8(v8h lo, v8h hi) {
    v16h a;
#pragma unroll
    for (int e = 0; e < 8; ++e) { a[e] = lo[e]; a[8 + e] = hi[e]; }
    return a;
}

// exchange with lane^16 : ds_swizzle_b32 SWAPX16 (xor=0x10, and=0x1f)
__device__ __forceinline__ float swapx16(float x) {
    int i = __builtin_bit_cast(int, x);
    i = __builtin_amdgcn_ds_swizzle(i, 0x401F);
    return __builtin_bit_cast(float, i);
}

// async byte-copy global->LDS, 16B per lane, tracked on ASYNCcnt
__device__ __forceinline__ void async_b128(unsigned lds_off, const void* gptr) {
    asm volatile("global_load_async_to_lds_b128 %0, %1, off"
                 :: "v"(lds_off), "v"(gptr) : "memory");
}
__device__ __forceinline__ void wait_async0() {
    asm volatile("s_wait_asynccnt 0x0" ::: "memory");
}

// Online-softmax over four 16x16 S^T tiles (64 keys x 16 q, log2 domain).
template <bool MASKED>
__device__ __forceinline__ void softmax_step(
    const v8f s[4], int thr, bool hi_half,
    float& mrow, float& lrow, v8f oc[8], v16h bp[2])
{
    v8f x[4];
#pragma unroll
    for (int t = 0; t < 4; ++t)
#pragma unroll
        for (int r = 0; r < 8; ++r)
            x[t][r] = (MASKED && (16 * t + r > thr)) ? -1e30f : s[t][r];

    float mloc = x[0][0];
#pragma unroll
    for (int t = 0; t < 4; ++t)
#pragma unroll
        for (int r = (t == 0); r < 8; ++r) mloc = fmaxf(mloc, x[t][r]);
    mloc = fmaxf(mloc, swapx16(mloc));
    const float mnew  = fmaxf(mrow, mloc);
    const float alpha = exp2f(mrow - mnew);

    v8f p[4];
    float ls = 0.0f;
#pragma unroll
    for (int t = 0; t < 4; ++t)
#pragma unroll
        for (int r = 0; r < 8; ++r) {
            p[t][r] = exp2f(x[t][r] - mnew);
            ls += p[t][r];
        }
    ls  += swapx16(ls);
    lrow = lrow * alpha + ls;
    mrow = mnew;
#pragma unroll
    for (int dc = 0; dc < 8; ++dc) oc[dc] = oc[dc] * alpha;

    // register transpose: P (C-layout f32) -> P^T B-fragments (f16)
    v8f ps[4];
#pragma unroll
    for (int t = 0; t < 4; ++t)
#pragma unroll
        for (int r = 0; r < 8; ++r) ps[t][r] = swapx16(p[t][r]);
#pragma unroll
    for (int f = 0; f < 2; ++f) {
        const v8f pt  = hi_half ? p [2*f+1] : p [2*f];
        const v8f pst = hi_half ? ps[2*f+1] : ps[2*f];
        const v8f lo  = hi_half ? pst : pt;   // keys 16h + 0..7
        const v8f hi  = hi_half ? pt  : pst;  // keys 16h + 8..15
#pragma unroll
        for (int r = 0; r < 8; ++r) {
            bp[f][r]     = (_Float16)lo[r];
            bp[f][8 + r] = (_Float16)hi[r];
        }
    }
}

// ---------------- pre-pass: K fp32 -> f16 [kvh][s][d] ----------------
__global__ __launch_bounds__(128) void cvt_k(const float* __restrict__ k,
                                             _Float16* __restrict__ k16) {
    const int h = blockIdx.y, d = threadIdx.x;
#pragma unroll
    for (int j = 0; j < 4; ++j) {
        const int s = blockIdx.x * 4 + j;
        k16[((size_t)h * SEQ_LEN + s) * HD + d] =
            (_Float16)k[((size_t)s * NUM_KV + h) * HD + d];
    }
}

// -------- pre-pass: V fp32 -> f16 transposed [kvh][d][s] (LDS tile) --------
__global__ __launch_bounds__(256) void cvt_v(const float* __restrict__ v,
                                             _Float16* __restrict__ vt) {
    __shared__ __align__(16) _Float16 lt[HD * 72];   // [d][s_local], pad 72
    const int s0 = blockIdx.x * 64, h = blockIdx.y, tid = threadIdx.x;
#pragma unroll
    for (int j = 0; j < 32; ++j) {                   // coalesced read over d
        const int e = tid + 256 * j;                 // 0..8191
        const int sl = e >> 7, d = e & 127;
        lt[d * 72 + sl] =
            (_Float16)v[((size_t)(s0 + sl) * NUM_KV + h) * HD + d];
    }
    __syncthreads();
#pragma unroll
    for (int j = 0; j < 8; ++j) {                    // coalesced v4h write
        const int e = tid + 256 * j;                 // 0..2047
        const int d = e >> 4, sl4 = (e & 15) * 4;
        *reinterpret_cast<v4h*>(
            &vt[((size_t)h * HD + d) * SEQ_LEN + s0 + sl4]) =
            *reinterpret_cast<const v4h*>(&lt[d * 72 + sl4]);
    }
}

// ---------------- fast path: async-staged hot loop ----------------
__global__ __launch_bounds__(128) void fa_fwd_async(
    const float* __restrict__ q, const _Float16* __restrict__ k16,
    const _Float16* __restrict__ vt, float* __restrict__ out)
{
    __shared__ __align__(16) _Float16 lk [BC * KSTR];   // K  tile [key][d]
    __shared__ __align__(16) _Float16 lvt[HD * VSTR];   // V^T tile [d][key]

    const int tid  = threadIdx.x;
    const int wave = tid >> 5;
    const int lane = tid & 31;
    const int col  = lane & 15;
    const int half = lane >> 4;
    const bool hi_half = (half != 0);

    const int qblk = blockIdx.x;
    const int head = blockIdx.y;
    const int kvh  = head >> 2;

    const int q0 = qblk * BR;
    const int qg = q0 + wave * 16 + col;

    // per-lane async-copy geometry (constant across iterations)
    const int krow = wave * 16 + (lane >> 4);        // +2 per instr slot
    const int kseg = (lane & 15) * 8;                // halves
    const int vrow = wave * 32 + (lane >> 3);        // +4 per instr slot
    const int vseg = (lane & 7) * 8;                 // halves
    const _Float16* kbase = k16 + (size_t)kvh * SEQ_LEN * HD;
    const _Float16* vbase = vt  + (size_t)kvh * HD * SEQ_LEN;

    // ---- Q^T B-fragments (pre-scaled by scale*log2e) ----
    v16h bq[4];
    {
        const float sc = ATT_SCALE * LOG2E;
        const float* qrow = q + ((size_t)qg * NUM_HEADS + head) * HD;
#pragma unroll
        for (int kk = 0; kk < 4; ++kk) {
            const float4* p4 =
                reinterpret_cast<const float4*>(qrow + 32 * kk + 16 * half);
#pragma unroll
            for (int i = 0; i < 4; ++i) {
                float4 f = p4[i];
                bq[kk][4*i+0] = (_Float16)(f.x * sc);
                bq[kk][4*i+1] = (_Float16)(f.y * sc);
                bq[kk][4*i+2] = (_Float16)(f.z * sc);
                bq[kk][4*i+3] = (_Float16)(f.w * sc);
            }
        }
    }

    v8f   oc[8] = {};
    float mrow = -1e30f, lrow = 0.0f;
    const int nkb = qblk + 1;

    // stage tile 0
#pragma unroll
    for (int i = 0; i < 8; ++i) {
        const int kr = krow + 2 * i;
        async_b128((unsigned)(size_t)&lk[kr * KSTR + kseg],
                   kbase + (size_t)kr * HD + kseg);
        const int vr = vrow + 4 * i;
        async_b128((unsigned)(size_t)&lvt[vr * VSTR + vseg],
                   vbase + (size_t)vr * SEQ_LEN + vseg);
    }

    for (int ib = 0; ib < nkb; ++ib) {
        const int kb = ib * BC;
        wait_async0();          // this wave's copies of tile ib complete
        __syncthreads();        // all waves' copies visible

        // ---- S^T = K * Q^T : four independent 16(key)x16(q) chains ----
        // kk-outer / t-inner: each round issues 8 loads feeding 4
        // independent WMMA chains -> partial dscnt waits, no serialization.
        v8f st[4] = {};
#pragma unroll
        for (int kk = 0; kk < 4; ++kk) {
#pragma unroll
            for (int t = 0; t < 4; ++t) {
                const int key = 16 * t + col;
                const v8h c0 = *reinterpret_cast<const v8h*>(
                    &lk[key * KSTR + 32 * kk + 16 * half]);
                const v8h c1 = *reinterpret_cast<const v8h*>(
                    &lk[key * KSTR + 32 * kk + 16 * half + 8]);
                st[t] = __builtin_amdgcn_wmma_f32_16x16x32_f16(
                    false, join8(c0, c1), false, bq[kk],
                    (short)0, st[t], false, false);
            }
        }

        // ---- online softmax; only the last (diagonal) block masks ----
        v16h bp[2];
        if (ib == nkb - 1) {
            softmax_step<true >(st, qg - kb - 8 * half, hi_half,
                                mrow, lrow, oc, bp);
        } else {
            softmax_step<false>(st, 0, hi_half, mrow, lrow, oc, bp);
        }

        // ---- O^T += V^T * P^T : 8 d-chunks, K-depth 64 (2 calls) ----
#pragma unroll
        for (int dc = 0; dc < 8; ++dc) {
            const int d = 16 * dc + col;
            const v8h a00 = *reinterpret_cast<const v8h*>(
                &lvt[d * VSTR + 16 * half]);
            const v8h a01 = *reinterpret_cast<const v8h*>(
                &lvt[d * VSTR + 16 * half + 8]);
            oc[dc] = __builtin_amdgcn_wmma_f32_16x16x32_f16(
                false, join8(a00, a01), false, bp[0],
                (short)0, oc[dc], false, false);
            const v8h a10 = *reinterpret_cast<const v8h*>(
                &lvt[d * VSTR + 32 + 16 * half]);
            const v8h a11 = *reinterpret_cast<const v8h*>(
                &lvt[d * VSTR + 40 + 16 * half]);
            oc[dc] = __builtin_amdgcn_wmma_f32_16x16x32_f16(
                false, join8(a10, a11), false, bp[1],
                (short)0, oc[dc], false, false);
        }
        __syncthreads();        // everyone done reading tile ib

        if (ib + 1 < nkb) {     // stage tile ib+1 (overlaps other blocks)
            const size_t kofs = (size_t)(kb + BC) * HD;
            const size_t vofs = (size_t)(kb + BC);
#pragma unroll
            for (int i = 0; i < 8; ++i) {
                const int kr = krow + 2 * i;
                async_b128((unsigned)(size_t)&lk[kr * KSTR + kseg],
                           kbase + kofs + (size_t)kr * HD + kseg);
                const int vr = vrow + 4 * i;
                async_b128((unsigned)(size_t)&lvt[vr * VSTR + vseg],
                           vbase + vofs + (size_t)vr * SEQ_LEN + vseg);
            }
        }
    }

    // ---- epilogue: O = O^T / l ----
    const float inv = 1.0f / lrow;
    float* orow = out + (size_t)qg * (NUM_HEADS * HD) + head * HD;
#pragma unroll
    for (int dc = 0; dc < 8; ++dc)
#pragma unroll
        for (int r = 0; r < 8; ++r)
            orow[16 * dc + 8 * half + r] = oc[dc][r] * inv;
}

// ---------------- fallback: in-loop conversion ----------------
__global__ __launch_bounds__(128) void fa_fwd_conv(
    const float* __restrict__ q, const float* __restrict__ k,
    const float* __restrict__ v, float* __restrict__ out)
{
    __shared__ __align__(16) _Float16 lk [BC * KSTR];
    __shared__ __align__(16) _Float16 lvt[HD * VSTR];

    const int tid  = threadIdx.x;
    const int lane = tid & 31;
    const int col  = lane & 15;
    const int half = lane >> 4;
    const bool hi_half = (half != 0);

    const int qblk = blockIdx.x;
    const int head = blockIdx.y;
    const int kvh  = head >> 2;

    const int q0 = qblk * BR;
    const int qg = q0 + (tid >> 5) * 16 + col;

    v16h bq[4];
    {
        const float sc = ATT_SCALE * LOG2E;
        const float* qrow = q + ((size_t)qg * NUM_HEADS + head) * HD;
#pragma unroll
        for (int kk = 0; kk < 4; ++kk) {
            const float4* p4 =
                reinterpret_cast<const float4*>(qrow + 32 * kk + 16 * half);
#pragma unroll
            for (int i = 0; i < 4; ++i) {
                float4 f = p4[i];
                bq[kk][4*i+0] = (_Float16)(f.x * sc);
                bq[kk][4*i+1] = (_Float16)(f.y * sc);
                bq[kk][4*i+2] = (_Float16)(f.z * sc);
                bq[kk][4*i+3] = (_Float16)(f.w * sc);
            }
        }
    }

    v8f   oc[8] = {};
    float mrow = -1e30f, lrow = 0.0f;
    const int nkb = qblk + 1;

    for (int ib = 0; ib < nkb; ++ib) {
        const int kb = ib * BC;
#pragma unroll
        for (int i = 0; i < 16; ++i) {
            const int idx = tid + 128 * i;
            const int key = idx >> 5;
            const int dv  = (idx & 31) << 2;
            const size_t g = ((size_t)(kb + key) * NUM_KV + kvh) * HD + dv;
            const float4 kf = *reinterpret_cast<const float4*>(k + g);
            const float4 vf = *reinterpret_cast<const float4*>(v + g);
            v4h kh;
            kh[0] = (_Float16)kf.x; kh[1] = (_Float16)kf.y;
            kh[2] = (_Float16)kf.z; kh[3] = (_Float16)kf.w;
            *reinterpret_cast<v4h*>(&lk[key * KSTR + dv]) = kh;
            lvt[(dv + 0) * VSTR + key] = (_Float16)vf.x;
            lvt[(dv + 1) * VSTR + key] = (_Float16)vf.y;
            lvt[(dv + 2) * VSTR + key] = (_Float16)vf.z;
            lvt[(dv + 3) * VSTR + key] = (_Float16)vf.w;
        }
        __syncthreads();

        v8f st[4] = {};
#pragma unroll
        for (int kk = 0; kk < 4; ++kk) {
#pragma unroll
            for (int t = 0; t < 4; ++t) {
                const int key = 16 * t + col;
                const v8h c0 = *reinterpret_cast<const v8h*>(
                    &lk[key * KSTR + 32 * kk + 16 * half]);
                const v8h c1 = *reinterpret_cast<const v8h*>(
                    &lk[key * KSTR + 32 * kk + 16 * half + 8]);
                st[t] = __builtin_amdgcn_wmma_f32_16x16x32_f16(
                    false, join8(c0, c1), false, bq[kk],
                    (short)0, st[t], false, false);
            }
        }

        v16h bp[2];
        if (ib == nkb - 1) {
            softmax_step<true >(st, qg - kb - 8 * half, hi_half,
                                mrow, lrow, oc, bp);
        } else {
            softmax_step<false>(st, 0, hi_half, mrow, lrow, oc, bp);
        }

#pragma unroll
        for (int dc = 0; dc < 8; ++dc) {
            const int d = 16 * dc + col;
            const v8h a00 = *reinterpret_cast<const v8h*>(
                &lvt[d * VSTR + 16 * half]);
            const v8h a01 = *reinterpret_cast<const v8h*>(
                &lvt[d * VSTR + 16 * half + 8]);
            oc[dc] = __builtin_amdgcn_wmma_f32_16x16x32_f16(
                false, join8(a00, a01), false, bp[0],
                (short)0, oc[dc], false, false);
            const v8h a10 = *reinterpret_cast<const v8h*>(
                &lvt[d * VSTR + 32 + 16 * half]);
            const v8h a11 = *reinterpret_cast<const v8h*>(
                &lvt[d * VSTR + 40 + 16 * half]);
            oc[dc] = __builtin_amdgcn_wmma_f32_16x16x32_f16(
                false, join8(a10, a11), false, bp[1],
                (short)0, oc[dc], false, false);
        }
        __syncthreads();
    }

    const float inv = 1.0f / lrow;
    float* orow = out + (size_t)qg * (NUM_HEADS * HD) + head * HD;
#pragma unroll
    for (int dc = 0; dc < 8; ++dc)
#pragma unroll
        for (int r = 0; r < 8; ++r)
            orow[16 * dc + 8 * half + r] = oc[dc][r] * inv;
}

extern "C" void kernel_launch(void* const* d_in, const int* in_sizes, int n_in,
                              void* d_out, int out_size, void* d_ws, size_t ws_size,
                              hipStream_t stream) {
    (void)in_sizes; (void)n_in; (void)out_size;
    const float* q = (const float*)d_in[0];
    const float* k = (const float*)d_in[1];
    const float* v = (const float*)d_in[2];
    float* out = (float*)d_out;
    dim3 grid(SEQ_LEN / BR, NUM_HEADS);
    dim3 block(128);

    const size_t ws_need =
        (size_t)2 * NUM_KV * SEQ_LEN * HD * sizeof(_Float16);   // 8 MB
    if (ws_size >= ws_need && d_ws) {
        _Float16* k16 = (_Float16*)d_ws;
        _Float16* vt  = k16 + (size_t)NUM_KV * SEQ_LEN * HD;
        cvt_k<<<dim3(SEQ_LEN / 4, NUM_KV), 128, 0, stream>>>(k, k16);
        cvt_v<<<dim3(SEQ_LEN / 64, NUM_KV), 256, 0, stream>>>(v, vt);
        fa_fwd_async<<<grid, block, 0, stream>>>(q, k16, vt, out);
    } else {
        fa_fwd_conv<<<grid, block, 0, stream>>>(q, k, v, out);
    }
}
